// DpnnBlock_22574348107972
// MI455X (gfx1250) — compile-verified, matrix operand order checked by reading, not run on hardware
//
#include <hip/hip_runtime.h>
#include <hip/hip_bf16.h>

// ---------------------------------------------------------------------------
// Dual-path RNN block (DPRNN) for MI455X / gfx1250, wave32 + WMMA f16->f32.
// B=8, L=200, W=64, C=H=128.  All GEMMs have K=128, M=102400.
// ---------------------------------------------------------------------------

#define Bb 8
#define Ll 200
#define Ww 64
#define Cc 128
#define Hh 128
#define MROWS (Bb * Ll * Ww)          // 102400
#define LWC   (Ll * Ww * Cc)          // 1638400 (== W*L*C)
#define SEQPB 16                      // sequences per GRU block (one 16-row M tile)

typedef __attribute__((ext_vector_type(16))) _Float16 v16h;
typedef __attribute__((ext_vector_type(8)))  float    v8f;

// Load a 16x32 f16 A/B fragment from an LDS tile with leading dim 128.
// Caller passes base = tile + row*128 + kc*32 + (lane>>4)*8 where
// row = 16*tile_idx + (lane&15).  Per ISA 7.12.2: VGPR v<4 holds K=2v,2v+1;
// v>=4 holds K=16+2(v-4)..  (+8 per lane-half): element j maps to
// offset j (j<8) or j+8 (j>=8) -> two aligned 16B chunks (ds_load_b128 x2).
__device__ inline v16h frag_ld(const _Float16* base) {
  v16h f;
#pragma unroll
  for (int j = 0; j < 8; ++j) f[j] = base[j];
#pragma unroll
  for (int j = 8; j < 16; ++j) f[j] = base[j + 8];
  return f;
}

// ---------------------------------------------------------------------------
// Generic GEMM: Out[M,Ntot] = A[M,128] * W[128,Ntot] + bias   (f32 in/out,
// f16 WMMA inside).  Block = 256 threads (8 waves), tile = 64 x 128.
// grid.x = M/64, grid.y = Ntot/128.
// ---------------------------------------------------------------------------
__global__ __launch_bounds__(256) void gemm_bias_wmma(
    const float* __restrict__ A, const float* __restrict__ W,
    const float* __restrict__ bias, float* __restrict__ Out, int Ntot) {
  __shared__ _Float16 a16[64 * 128];   // A tile, row major
  __shared__ _Float16 bT[128 * 128];   // bT[n][k] = W[k][n0+n]

  const int  tid    = threadIdx.x;
  const long m_base = (long)blockIdx.x * 64;
  const int  n0     = blockIdx.y * 128;

  // A tile: 64x128 f32 -> f16 (contiguous rows in global)
  {
    const float4* Ap = (const float4*)(A + m_base * 128);
    for (int i = tid; i < (64 * 128) / 4; i += 256) {
      float4 v = Ap[i];
      int o = i * 4;
      a16[o + 0] = (_Float16)v.x; a16[o + 1] = (_Float16)v.y;
      a16[o + 2] = (_Float16)v.z; a16[o + 3] = (_Float16)v.w;
    }
  }
  // W panel, transposed so (k,k+1) pairs are contiguous for B fragments
  for (int i = tid; i < 128 * 32; i += 256) {
    int k  = i >> 5;
    int nq = (i & 31) * 4;
    float4 v = *(const float4*)(W + (long)k * Ntot + n0 + nq);
    bT[(nq + 0) * 128 + k] = (_Float16)v.x;
    bT[(nq + 1) * 128 + k] = (_Float16)v.y;
    bT[(nq + 2) * 128 + k] = (_Float16)v.z;
    bT[(nq + 3) * 128 + k] = (_Float16)v.w;
  }
  __syncthreads();

  const int wv = tid >> 5, lane = tid & 31;
  const int half_ = lane >> 4, lrow = lane & 15;
  const int nt = wv;                       // each wave owns one 16-col strip
  const int ncol = n0 + nt * 16 + lrow;
  const float bv = bias[ncol];

  for (int mt = 0; mt < 4; ++mt) {
    v8f acc = {};
#pragma unroll
    for (int kc = 0; kc < 4; ++kc) {
      v16h af = frag_ld(a16 + (mt * 16 + lrow) * 128 + kc * 32 + half_ * 8);
      v16h bf = frag_ld(bT  + (nt * 16 + lrow) * 128 + kc * 32 + half_ * 8);
      acc = __builtin_amdgcn_wmma_f32_16x16x32_f16(
          false, af, false, bf, (short)0, acc, false, false);
    }
#pragma unroll
    for (int r = 0; r < 8; ++r) {
      long m = m_base + mt * 16 + r + 8 * half_;
      Out[m * Ntot + ncol] = acc[r] + bv;
    }
  }
}

// ---------------------------------------------------------------------------
// GRU scan.  xp[N,T,384] precomputed (x@Wi + b).  Wh[128,384] kept f16 in
// LDS (transposed, B-fragment friendly).  SEQPB=16 sequences per 256-thread
// block (8 waves, one 16-row M tile); hidden state f32 + f16 mirror resident
// in LDS for the whole scan.  Step-t xp is loaded into registers *before*
// the WMMA phase so HBM latency overlaps the matmul instead of sitting on
// the sequential critical path; t+1 row is prefetched (global_prefetch_b8).
// ---------------------------------------------------------------------------
__global__ __launch_bounds__(256) void gru_scan_wmma(
    const float* __restrict__ xp, const float* __restrict__ Wh,
    float* __restrict__ ys, int T) {
  extern __shared__ char smem[];
  _Float16* whT = (_Float16*)smem;                          // [384][128] 96KB
  float*    hf  = (float*)(smem + 98304);                   // [16][128]   8KB
  _Float16* h16 = (_Float16*)(smem + 98304 + 8192);         // [16][128]   4KB
  float*    hwh = (float*)(smem + 98304 + 8192 + 4096);     // [16][384]  24KB

  const int  tid  = threadIdx.x;
  const long seq0 = (long)blockIdx.x * SEQPB;

  // Wh -> whT (f16), vectorized along n
  for (int i = tid; i < 12288; i += 256) {    // 128*384/4
    int k = i / 96;
    int n = (i % 96) * 4;
    float4 v = *(const float4*)(Wh + k * 384 + n);
    whT[(n + 0) * 128 + k] = (_Float16)v.x;
    whT[(n + 1) * 128 + k] = (_Float16)v.y;
    whT[(n + 2) * 128 + k] = (_Float16)v.z;
    whT[(n + 3) * 128 + k] = (_Float16)v.w;
  }
  for (int i = tid; i < SEQPB * 128; i += 256) { hf[i] = 0.f; h16[i] = (_Float16)0.f; }
  __syncthreads();

  const int wv = tid >> 5, lane = tid & 31;
  const int half_ = lane >> 4, lrow = lane & 15;
  const int s  = tid >> 4;           // sequence 0..15 for gate phase
  const int hb = (tid & 15) * 8;     // 8 hidden elems per thread

  const float* xbase = xp + (seq0 + s) * (long)T * 384;

  for (int t = 0; t < T; ++t) {
    // ---- issue step-t xp loads now; they overlap the WMMA phase below ----
    const float* xr_t = xbase + (long)t * 384;
    float4 pz0 = *(const float4*)(xr_t + hb);
    float4 pz1 = *(const float4*)(xr_t + hb + 4);
    float4 pr0 = *(const float4*)(xr_t + 128 + hb);
    float4 pr1 = *(const float4*)(xr_t + 128 + hb + 4);
    float4 ph0 = *(const float4*)(xr_t + 256 + hb);
    float4 ph1 = *(const float4*)(xr_t + 256 + hb + 4);
    if (t + 1 < T) __builtin_prefetch(xr_t + 384 + hb, 0, 0);

    // -------- h16[16,128] x whT[128,384] -> hwh[16,384] --------
#pragma unroll
    for (int q = 0; q < 3; ++q) {
      int nt = wv * 3 + q;           // 24 N-tiles across 8 waves
      v8f acc = {};
#pragma unroll
      for (int kc = 0; kc < 4; ++kc) {
        v16h af = frag_ld(h16 + lrow * 128 + kc * 32 + half_ * 8);
        v16h bf = frag_ld(whT + (nt * 16 + lrow) * 128 + kc * 32 + half_ * 8);
        acc = __builtin_amdgcn_wmma_f32_16x16x32_f16(
            false, af, false, bf, (short)0, acc, false, false);
      }
#pragma unroll
      for (int r = 0; r < 8; ++r)
        hwh[(r + 8 * half_) * 384 + nt * 16 + lrow] = acc[r];
    }
    __syncthreads();

    // -------- gates (8 elems per thread) --------
    float xz[8], xr[8], xh[8];
    *(float4*)&xz[0] = pz0; *(float4*)&xz[4] = pz1;
    *(float4*)&xr[0] = pr0; *(float4*)&xr[4] = pr1;
    *(float4*)&xh[0] = ph0; *(float4*)&xh[4] = ph1;
    const float* hwrow = hwh + s * 384;
    float hn[8];
#pragma unroll
    for (int i = 0; i < 8; ++i) {
      int   hx = hb + i;
      float az = xz[i] + hwrow[hx];
      float ar = xr[i] + hwrow[128 + hx];
      float z  = 1.f / (1.f + __expf(-az));
      float r  = 1.f / (1.f + __expf(-ar));
      float hc = tanhf(xh[i] + r * hwrow[256 + hx]);
      float ho = hf[s * 128 + hx];
      hn[i] = z * ho + (1.f - z) * hc;
    }
    float* yrow = ys + ((seq0 + s) * (long)T + t) * 128 + hb;
    *(float4*)(yrow)     = *(const float4*)&hn[0];
    *(float4*)(yrow + 4) = *(const float4*)&hn[4];
#pragma unroll
    for (int i = 0; i < 8; ++i) {
      hf[s * 128 + hb + i]  = hn[i];
      h16[s * 128 + hb + i] = (_Float16)hn[i];
    }
    __syncthreads();   // h16/hf visible before next step's WMMA; hwh reusable
  }
}

// ---------------------------------------------------------------------------
// LayerNorm helpers (global LN over LWC elements per batch item)
// ---------------------------------------------------------------------------
__global__ void zero_stats(float* stats) {
  if (threadIdx.x < 16) stats[threadIdx.x] = 0.f;
}

__global__ __launch_bounds__(256) void reduce_stats(
    const float* __restrict__ d, float* __restrict__ stats) {
  __shared__ float ss[256], sq[256];
  long base = (long)blockIdx.x * 8192 + threadIdx.x * 4;
  float s = 0.f, q = 0.f;
#pragma unroll
  for (int i = 0; i < 8; ++i) {
    float4 v = *(const float4*)(d + base + i * 1024);
    s += v.x + v.y + v.z + v.w;
    q += v.x * v.x + v.y * v.y + v.z * v.z + v.w * v.w;
  }
  ss[threadIdx.x] = s; sq[threadIdx.x] = q;
  __syncthreads();
  for (int st = 128; st > 0; st >>= 1) {
    if (threadIdx.x < st) {
      ss[threadIdx.x] += ss[threadIdx.x + st];
      sq[threadIdx.x] += sq[threadIdx.x + st];
    }
    __syncthreads();
  }
  if (threadIdx.x == 0) {
    int b = blockIdx.x / 200;     // 1638400 / 8192 = 200 blocks per batch
    atomicAdd(&stats[2 * b + 0], ss[0]);
    atomicAdd(&stats[2 * b + 1], sq[0]);
  }
}

// intra = x + LN(d) * g + be    (all contiguous, same layout)
__global__ __launch_bounds__(256) void finalize_intra(
    const float* __restrict__ x, const float* __restrict__ d,
    const float* __restrict__ g, const float* __restrict__ be,
    const float* __restrict__ stats, float* __restrict__ intra) {
  long idx = (long)blockIdx.x * 256 + threadIdx.x;
  long fi  = idx * 4;
  int  b   = (int)(fi / LWC);
  long j   = fi % LWC;
  float m   = stats[2 * b] * (1.f / (float)LWC);
  float var = stats[2 * b + 1] * (1.f / (float)LWC) - m * m;
  float rs  = rsqrtf(var + 1e-8f);
  float4 dv = *(const float4*)(d + fi);
  float4 xv = *(const float4*)(x + fi);
  float4 gv = *(const float4*)(g + j);
  float4 bv = *(const float4*)(be + j);
  float4 o;
  o.x = xv.x + (dv.x - m) * rs * gv.x + bv.x;
  o.y = xv.y + (dv.y - m) * rs * gv.y + bv.y;
  o.z = xv.z + (dv.z - m) * rs * gv.z + bv.z;
  o.w = xv.w + (dv.w - m) * rs * gv.w + bv.w;
  *(float4*)(intra + fi) = o;
}

// ii[b,w,l,c] = intra[b,l,w,c]
__global__ __launch_bounds__(256) void transpose_lw(
    const float* __restrict__ in, float* __restrict__ out) {
  long idx = (long)blockIdx.x * 256 + threadIdx.x;   // float4 index over output
  int  c4  = (int)(idx & 31);
  long r   = idx >> 5;             // row over [B*W*L]
  int  l   = (int)(r % Ll);
  long bw  = r / Ll;
  int  w   = (int)(bw % Ww);
  int  b   = (int)(bw / Ww);
  const float4* ip = (const float4*)in + (((long)(b * Ll + l) * Ww + w) * 32 + c4);
  ((float4*)out)[idx] = *ip;
}

// out[b,l,w,c] = intra[b,l,w,c] + LN(d2)[b,w,l,c]*g + be   (d2 layout [B,W,L,C])
__global__ __launch_bounds__(256) void finalize_out(
    const float* __restrict__ intra, const float* __restrict__ d2,
    const float* __restrict__ g, const float* __restrict__ be,
    const float* __restrict__ stats, float* __restrict__ out) {
  long idx = (long)blockIdx.x * 256 + threadIdx.x;
  long fi  = idx * 4;                 // flat index in d2 layout
  int  b   = (int)(fi / LWC);
  long j   = fi % LWC;
  float m   = stats[2 * b] * (1.f / (float)LWC);
  float var = stats[2 * b + 1] * (1.f / (float)LWC) - m * m;
  float rs  = rsqrtf(var + 1e-8f);
  long rowd = fi / 128;               // row over [B*W, L]
  int  l    = (int)(rowd % Ll);
  long bw   = rowd / Ll;
  int  w    = (int)(bw % Ww);
  long ofi  = (((long)(b * Ll + l) * Ww + w) * 128) + (fi & 127);
  float4 dv = *(const float4*)(d2 + fi);
  float4 gv = *(const float4*)(g + j);
  float4 bv = *(const float4*)(be + j);
  float4 iv = *(const float4*)(intra + ofi);
  float4 o;
  o.x = iv.x + (dv.x - m) * rs * gv.x + bv.x;
  o.y = iv.y + (dv.y - m) * rs * gv.y + bv.y;
  o.z = iv.z + (dv.z - m) * rs * gv.z + bv.z;
  o.w = iv.w + (dv.w - m) * rs * gv.w + bv.w;
  *(float4*)(out + ofi) = o;
}

// ---------------------------------------------------------------------------
extern "C" void kernel_launch(void* const* d_in, const int* in_sizes, int n_in,
                              void* d_out, int out_size, void* d_ws, size_t ws_size,
                              hipStream_t stream) {
  const float* x        = (const float*)d_in[0];
  const float* W_in     = (const float*)d_in[1];
  const float* b_in     = (const float*)d_in[2];
  const float* intra_Wi = (const float*)d_in[3];
  const float* intra_Wh = (const float*)d_in[4];
  const float* intra_b  = (const float*)d_in[5];
  const float* intra_fW = (const float*)d_in[6];
  const float* intra_fb = (const float*)d_in[7];
  const float* intra_g  = (const float*)d_in[8];
  const float* intra_be = (const float*)d_in[9];
  const float* inter_Wi = (const float*)d_in[10];
  const float* inter_Wh = (const float*)d_in[11];
  const float* inter_b  = (const float*)d_in[12];
  const float* inter_fW = (const float*)d_in[13];
  const float* inter_fb = (const float*)d_in[14];
  const float* inter_g  = (const float*)d_in[15];
  const float* inter_be = (const float*)d_in[16];
  float* out = (float*)d_out;

  // workspace carve-up (floats)
  float* XX    = (float*)d_ws;                   // [102400,128] xx, later ii
  float* XP    = XX    + (long)MROWS * 128;      // [102400,384]
  float* G     = XP    + (long)MROWS * 384;      // [102400,128] GRU outputs
  float* Dd    = G     + (long)MROWS * 128;      // [102400,128] fc outputs
  float* INTRA = Dd    + (long)MROWS * 128;      // [102400,128]
  float* stats = INTRA + (long)MROWS * 128;      // 16 floats

  const int gruSmem = 98304 + 8192 + 4096 + SEQPB * 384 * 4;  // 135168 B
  hipFuncSetAttribute((const void*)gru_scan_wmma,
                      hipFuncAttributeMaxDynamicSharedMemorySize, gruSmem);

  dim3 blk(256);

  // 1) xx = x @ W_in + b_in
  gemm_bias_wmma<<<dim3(MROWS / 64, 1), blk, 0, stream>>>(x, W_in, b_in, XX, 128);
  // 2) xp = xx @ intra_Wi + intra_b
  gemm_bias_wmma<<<dim3(MROWS / 64, 3), blk, 0, stream>>>(XX, intra_Wi, intra_b, XP, 384);
  // 3) intra GRU scan: 1600 sequences of length 64 -> 100 blocks
  gru_scan_wmma<<<dim3((Bb * Ll) / SEQPB), blk, gruSmem, stream>>>(XP, intra_Wh, G, Ww);
  // 4) d = ig @ intra_fcW + intra_fcb
  gemm_bias_wmma<<<dim3(MROWS / 64, 1), blk, 0, stream>>>(G, intra_fW, intra_fb, Dd, 128);
  // 5) global LN stats over each batch's 1.64M elements
  zero_stats<<<dim3(1), dim3(64), 0, stream>>>(stats);
  reduce_stats<<<dim3(1600), blk, 0, stream>>>(Dd, stats);
  // 6) intra = x + LN(d)
  finalize_intra<<<dim3(12800), blk, 0, stream>>>(x, Dd, intra_g, intra_be, stats, INTRA);
  // 7) ii = transpose(intra, (0,2,1,3))  -> reuse XX
  transpose_lw<<<dim3(12800), blk, 0, stream>>>(INTRA, XX);
  // 8) xp = ii @ inter_Wi + inter_b
  gemm_bias_wmma<<<dim3(MROWS / 64, 3), blk, 0, stream>>>(XX, inter_Wi, inter_b, XP, 384);
  // 9) inter GRU scan: 512 sequences of length 200 -> 32 blocks
  gru_scan_wmma<<<dim3((Bb * Ww) / SEQPB), blk, gruSmem, stream>>>(XP, inter_Wh, G, Ll);
  // 10) d2 = og @ inter_fcW + inter_fcb
  gemm_bias_wmma<<<dim3(MROWS / 64, 1), blk, 0, stream>>>(G, inter_fW, inter_fb, Dd, 128);
  // 11) LN stats for d2
  zero_stats<<<dim3(1), dim3(64), 0, stream>>>(stats);
  reduce_stats<<<dim3(1600), blk, 0, stream>>>(Dd, stats);
  // 12) out = intra + transpose(LN(d2))
  finalize_out<<<dim3(12800), blk, 0, stream>>>(INTRA, Dd, inter_g, inter_be, stats, out);
}
